// DeepseekMoE_56014963474753
// MI455X (gfx1250) — matrix operand order repaired and checked by reading, not compiled, not run on hardware
//
#include <hip/hip_runtime.h>

// ---------------- static problem config (mirrors reference) ----------------
#define T_TOK  2048
#define HDIM   2048
#define NEXP   32
#define TOPK   6
#define IDIM   1024
#define NGROUP 8
#define TOPKG  4
#define CAP    1024
#define RSCALE 2.5f

// ---------------- CDNA5 WMMA types ----------------
typedef __attribute__((ext_vector_type(16))) __bf16 v16bf;
typedef __attribute__((ext_vector_type(8)))  float  v8f;

union Frag16 { v16bf v; unsigned u[8]; uint4 q[2]; };

__device__ __forceinline__ unsigned short f2bfbits(float f) {
  unsigned u = __builtin_bit_cast(unsigned, f);
  unsigned r = (u + 0x7fffu + ((u >> 16) & 1u)) >> 16;
  return (unsigned short)r;
}
__device__ __forceinline__ unsigned pack2(float lo, float hi) {
  return (unsigned)f2bfbits(lo) | ((unsigned)f2bfbits(hi) << 16);
}
__device__ __forceinline__ float bf2f(unsigned short s) {
  unsigned u = ((unsigned)s) << 16;
  return __builtin_bit_cast(float, u);
}

__device__ __forceinline__ v8f wmma_bf16(v16bf a, v16bf b, v8f c) {
  return __builtin_amdgcn_wmma_f32_16x16x32_bf16(
      /*neg_a=*/false, a, /*neg_b=*/false, b,
      /*c_mod=*/(short)0, c, /*reuse_a=*/false, /*reuse_b=*/false);
}

// fast sigmoid: v_exp + v_add + v_rcp + (mul) — avoids the exact-division chain
__device__ __forceinline__ float fast_sigmoid(float x) {
  return __builtin_amdgcn_rcpf(1.f + __expf(-x));
}
__device__ __forceinline__ float silu(float g) { return g * fast_sigmoid(g); }

// B fragment (32x16 bf16): lane L holds column N=L%16, 16 K values in table order.
// Bk points at row k0 of a row-major bf16 matrix; prefetch next k-tile.
__device__ __forceinline__ v16bf load_b_bf16(const unsigned short* __restrict__ Bk,
                                             int ldb, int n0) {
  const int lane = threadIdx.x & 31;
  const unsigned short* row = Bk + (size_t)lane * ldb + n0;
  __builtin_prefetch((const char*)row + (size_t)64 * ldb, 0, 1);  // k0+32 rows
  const unsigned short* col = Bk + n0 + (lane & 15);
  const int hb = (lane >> 4) * 8;
  Frag16 f;
#pragma unroll
  for (int i = 0; i < 8; ++i) {
    const int kk = (i < 4 ? 2 * i : 2 * i + 8) + hb;
    f.u[i] = (unsigned)col[(size_t)kk * ldb] |
             ((unsigned)col[(size_t)(kk + 1) * ldb] << 16);
  }
  return f.v;
}

// A fragment (16x32 bf16) from LDS tile of packed pairs (16 rows x 16 dwords).
__device__ __forceinline__ v16bf load_a_lds(const unsigned* __restrict__ lds_a) {
  const int lane = threadIdx.x & 31;
  const int M = lane & 15;
  const int hbase = (lane >> 4) ? 4 : 0;
  Frag16 f;
#pragma unroll
  for (int i = 0; i < 8; ++i) {
    const int j = (i < 4 ? i : i + 4) + hbase;
    f.u[i] = lds_a[M * 16 + j];
  }
  return f.v;
}

// ---------------- elementwise f32 -> bf16 (vectorized, RNE) ----------------
__global__ void moe_cvt_kernel(const float* __restrict__ src,
                               unsigned short* __restrict__ dst, size_t n4) {
  size_t i = (size_t)blockIdx.x * blockDim.x + threadIdx.x;
  if (i >= n4) return;
  const float4 v = ((const float4*)src)[i];
  uint2 o;
  o.x = pack2(v.x, v.y);
  o.y = pack2(v.z, v.w);
  ((uint2*)dst)[i] = o;
}

__global__ void moe_zero_counts_kernel(int* __restrict__ counts) {
  if (threadIdx.x < NEXP) counts[threadIdx.x] = 0;
}

// ---------------- grouped top-k routing (one wave32 per token) -------------
__global__ __launch_bounds__(32)
void moe_route_kernel(const float* __restrict__ x, const float* __restrict__ gw,
                      const float* __restrict__ bias,
                      int* __restrict__ topk_idx, float* __restrict__ topk_w,
                      int* __restrict__ counts, int* __restrict__ pair_of_slot) {
  const int t = blockIdx.x;
  const int e = threadIdx.x;  // one expert per lane, E == wave32
  const float* xr = x + (size_t)t * HDIM;
  const float* gr = gw + (size_t)e * HDIM;
  float acc = 0.f;
#pragma unroll 4
  for (int i = 0; i < HDIM; ++i) acc += xr[i] * gr[i];
  const float score = fast_sigmoid(acc);
  float sfc = score + bias[e];

  // top-2 sum within each group of 4 experts
  const int gbase = e & ~3;
  float a0 = __shfl(sfc, gbase + 0, 32);
  float a1 = __shfl(sfc, gbase + 1, 32);
  float a2 = __shfl(sfc, gbase + 2, 32);
  float a3 = __shfl(sfc, gbase + 3, 32);
  float hi1 = fmaxf(a0, a1), lo1 = fminf(a0, a1);
  float hi2 = fmaxf(a2, a3), lo2 = fminf(a2, a3);
  float m1 = fmaxf(hi1, hi2);
  float m2 = fmaxf(fminf(hi1, hi2), (hi1 >= hi2) ? lo1 : lo2);
  float gscore = m1 + m2;

  float gs[NGROUP];
#pragma unroll
  for (int g = 0; g < NGROUP; ++g) gs[g] = __shfl(gscore, g * 4, 32);
  unsigned gmask = 0;
#pragma unroll
  for (int r = 0; r < TOPKG; ++r) {
    int best = 0; float bv = gs[0];
#pragma unroll
    for (int g = 1; g < NGROUP; ++g)
      if (gs[g] > bv) { bv = gs[g]; best = g; }
    gmask |= 1u << best;
    gs[best] = -3.4e38f;
  }
  float masked = ((gmask >> (e >> 2)) & 1u) ? sfc : 0.f;

  int kidx[TOPK];
  float kw[TOPK];
  float wsum = 0.f;
#pragma unroll
  for (int k = 0; k < TOPK; ++k) {
    float v = masked;
#pragma unroll
    for (int off = 16; off > 0; off >>= 1)
      v = fmaxf(v, __shfl_xor(v, off, 32));
    unsigned long long b = __ballot(masked == v);
    int sel = __ffsll(b) - 1;
    kidx[k] = sel;
    float w = __shfl(score, sel, 32);
    kw[k] = w;
    wsum += w;
    if (e == sel) masked = -3.4e38f;
  }
  const float inv = RSCALE * __builtin_amdgcn_rcpf(wsum + 1e-20f);
  if (e == 0) {
#pragma unroll
    for (int k = 0; k < TOPK; ++k) {
      topk_idx[t * TOPK + k] = kidx[k];
      topk_w[t * TOPK + k] = kw[k] * inv;
      int pos = atomicAdd(&counts[kidx[k]], 1);
      pair_of_slot[kidx[k] * CAP + pos] = t * TOPK + k;
    }
  }
}

// ---------------- routed experts gate-up + SiLU*mul ------------------------
// block: 128 thr (4 waves); tile 32 rows x (4 waves * 32 cols of I)
__global__ __launch_bounds__(128)
void moe_expert_gu_kernel(const unsigned short* __restrict__ xb,
                          const unsigned short* __restrict__ w_gu,
                          const int* __restrict__ counts,
                          const int* __restrict__ pair_of_slot,
                          unsigned short* __restrict__ hbuf) {
  const int e = blockIdx.z;
  const int m0 = blockIdx.y * 32;
  const int cnt = counts[e];
  if (m0 >= cnt) return;
  const int wave = threadIdx.x >> 5;
  const int lane = threadIdx.x & 31;
  const int nb = blockIdx.x * 128 + wave * 32;

  __shared__ int s_tok[32];
  __shared__ int s_pair[32];
  __shared__ unsigned s_a[32 * 16];
  const int* slots = pair_of_slot + e * CAP;
  if (threadIdx.x < 32) {
    int s = m0 + threadIdx.x;
    int pair = (s < cnt) ? slots[s] : 0;
    s_pair[threadIdx.x] = pair;
    s_tok[threadIdx.x] = pair / TOPK;
  }
  __syncthreads();

  const unsigned short* Bg = w_gu + (size_t)e * HDIM * (2 * IDIM);
  const int ldb = 2 * IDIM;
  v8f g00{}, g01{}, u00{}, u01{};
  v8f g10{}, g11{}, u10{}, u11{};

  for (int k0 = 0; k0 < HDIM; k0 += 32) {
    int idx = threadIdx.x;
#pragma unroll
    for (int r = 0; r < 4; ++r, idx += 128) {
      const int m = idx >> 4, j = idx & 15;
      const unsigned* xr =
          (const unsigned*)xb + ((size_t)s_tok[m] * HDIM + k0) / 2;
      s_a[idx] = xr[j];
    }
    __syncthreads();
    v16bf a0 = load_a_lds(s_a);        // rows 0..15
    v16bf a1 = load_a_lds(s_a + 256);  // rows 16..31
    const unsigned short* Bk = Bg + (size_t)k0 * ldb;
    v16bf bg0 = load_b_bf16(Bk, ldb, nb);
    v16bf bg1 = load_b_bf16(Bk, ldb, nb + 16);
    v16bf bu0 = load_b_bf16(Bk, ldb, IDIM + nb);
    v16bf bu1 = load_b_bf16(Bk, ldb, IDIM + nb + 16);
    g00 = wmma_bf16(a0, bg0, g00);
    g01 = wmma_bf16(a0, bg1, g01);
    u00 = wmma_bf16(a0, bu0, u00);
    u01 = wmma_bf16(a0, bu1, u01);
    g10 = wmma_bf16(a1, bg0, g10);
    g11 = wmma_bf16(a1, bg1, g11);
    u10 = wmma_bf16(a1, bu0, u10);
    u11 = wmma_bf16(a1, bu1, u11);
    __syncthreads();
  }

  const int rowoff = (lane >> 4) ? 8 : 0;
  const int col = lane & 15;
#pragma unroll
  for (int r = 0; r < 8; ++r) {
    int row = rowoff + r;
    if (m0 + row < cnt) {
      const int pair = s_pair[row];
      hbuf[(size_t)pair * IDIM + nb + col]      = f2bfbits(silu(g00[r]) * u00[r]);
      hbuf[(size_t)pair * IDIM + nb + 16 + col] = f2bfbits(silu(g01[r]) * u01[r]);
    }
    row += 16;
    if (m0 + row < cnt) {
      const int pair = s_pair[row];
      hbuf[(size_t)pair * IDIM + nb + col]      = f2bfbits(silu(g10[r]) * u10[r]);
      hbuf[(size_t)pair * IDIM + nb + 16 + col] = f2bfbits(silu(g11[r]) * u11[r]);
    }
  }
}

// ---------------- routed experts down-proj ---------------------------------
__global__ __launch_bounds__(128)
void moe_expert_down_kernel(const unsigned short* __restrict__ hbuf,
                            const unsigned short* __restrict__ w_down,
                            const int* __restrict__ counts,
                            const int* __restrict__ pair_of_slot,
                            unsigned short* __restrict__ ybuf) {
  const int e = blockIdx.z;
  const int m0 = blockIdx.y * 32;
  const int cnt = counts[e];
  if (m0 >= cnt) return;
  const int wave = threadIdx.x >> 5;
  const int lane = threadIdx.x & 31;
  const int nb = blockIdx.x * 128 + wave * 32;

  __shared__ int s_pair[32];
  __shared__ unsigned s_a[32 * 16];
  const int* slots = pair_of_slot + e * CAP;
  if (threadIdx.x < 32) {
    int s = m0 + threadIdx.x;
    s_pair[threadIdx.x] = (s < cnt) ? slots[s] : 0;
  }
  __syncthreads();

  const unsigned short* Bd = w_down + (size_t)e * IDIM * HDIM;
  v8f c00{}, c01{}, c10{}, c11{};

  for (int k0 = 0; k0 < IDIM; k0 += 32) {
    int idx = threadIdx.x;
#pragma unroll
    for (int r = 0; r < 4; ++r, idx += 128) {
      const int m = idx >> 4, j = idx & 15;
      const unsigned* hr =
          (const unsigned*)hbuf + ((size_t)s_pair[m] * IDIM + k0) / 2;
      s_a[idx] = hr[j];
    }
    __syncthreads();
    v16bf a0 = load_a_lds(s_a);
    v16bf a1 = load_a_lds(s_a + 256);
    const unsigned short* Bk = Bd + (size_t)k0 * HDIM;
    v16bf b0 = load_b_bf16(Bk, HDIM, nb);
    v16bf b1 = load_b_bf16(Bk, HDIM, nb + 16);
    c00 = wmma_bf16(a0, b0, c00);
    c01 = wmma_bf16(a0, b1, c01);
    c10 = wmma_bf16(a1, b0, c10);
    c11 = wmma_bf16(a1, b1, c11);
    __syncthreads();
  }

  const int rowoff = (lane >> 4) ? 8 : 0;
  const int col = lane & 15;
#pragma unroll
  for (int r = 0; r < 8; ++r) {
    int row = rowoff + r;
    if (m0 + row < cnt) {
      const int pair = s_pair[row];
      ybuf[(size_t)pair * HDIM + nb + col]      = f2bfbits(c00[r]);
      ybuf[(size_t)pair * HDIM + nb + 16 + col] = f2bfbits(c01[r]);
    }
    row += 16;
    if (m0 + row < cnt) {
      const int pair = s_pair[row];
      ybuf[(size_t)pair * HDIM + nb + col]      = f2bfbits(c10[r]);
      ybuf[(size_t)pair * HDIM + nb + 16 + col] = f2bfbits(c11[r]);
    }
  }
}

// ---------------- shared MLP gate-up + SiLU*mul ----------------------------
__global__ __launch_bounds__(128)
void moe_shared_gu_kernel(const unsigned short* __restrict__ xb,
                          const unsigned short* __restrict__ w_sgu,
                          unsigned short* __restrict__ sbuf) {
  const int t0 = blockIdx.y * 32;
  const int wave = threadIdx.x >> 5;
  const int lane = threadIdx.x & 31;
  const int nb = blockIdx.x * 128 + wave * 32;  // in [0, 2*IDIM)
  const int IS = 2 * IDIM;
  const int ldb = 2 * IS;  // w_sgu: [H, 4096]

  __shared__ unsigned s_a[32 * 16];
  v8f g00{}, g01{}, u00{}, u01{};
  v8f g10{}, g11{}, u10{}, u11{};

  for (int k0 = 0; k0 < HDIM; k0 += 32) {
    int idx = threadIdx.x;
#pragma unroll
    for (int r = 0; r < 4; ++r, idx += 128) {
      const int m = idx >> 4, j = idx & 15;
      const unsigned* xr =
          (const unsigned*)xb + ((size_t)(t0 + m) * HDIM + k0) / 2;
      s_a[idx] = xr[j];
    }
    __syncthreads();
    v16bf a0 = load_a_lds(s_a);
    v16bf a1 = load_a_lds(s_a + 256);
    const unsigned short* Bk = w_sgu + (size_t)k0 * ldb;
    v16bf bg0 = load_b_bf16(Bk, ldb, nb);
    v16bf bg1 = load_b_bf16(Bk, ldb, nb + 16);
    v16bf bu0 = load_b_bf16(Bk, ldb, IS + nb);
    v16bf bu1 = load_b_bf16(Bk, ldb, IS + nb + 16);
    g00 = wmma_bf16(a0, bg0, g00);
    g01 = wmma_bf16(a0, bg1, g01);
    u00 = wmma_bf16(a0, bu0, u00);
    u01 = wmma_bf16(a0, bu1, u01);
    g10 = wmma_bf16(a1, bg0, g10);
    g11 = wmma_bf16(a1, bg1, g11);
    u10 = wmma_bf16(a1, bu0, u10);
    u11 = wmma_bf16(a1, bu1, u11);
    __syncthreads();
  }

  const int rowoff = (lane >> 4) ? 8 : 0;
  const int col = lane & 15;
#pragma unroll
  for (int r = 0; r < 8; ++r) {
    const int t = t0 + rowoff + r;
    sbuf[(size_t)t * IS + nb + col]             = f2bfbits(silu(g00[r]) * u00[r]);
    sbuf[(size_t)t * IS + nb + 16 + col]        = f2bfbits(silu(g01[r]) * u01[r]);
    sbuf[(size_t)(t + 16) * IS + nb + col]      = f2bfbits(silu(g10[r]) * u10[r]);
    sbuf[(size_t)(t + 16) * IS + nb + 16 + col] = f2bfbits(silu(g11[r]) * u11[r]);
  }
}

// ---------------- shared MLP down-proj (writes f32 out) --------------------
__global__ __launch_bounds__(128)
void moe_shared_down_kernel(const unsigned short* __restrict__ sbuf,
                            const unsigned short* __restrict__ w_sdown,
                            float* __restrict__ out) {
  const int t0 = blockIdx.y * 32;
  const int wave = threadIdx.x >> 5;
  const int lane = threadIdx.x & 31;
  const int nb = blockIdx.x * 128 + wave * 32;
  const int IS = 2 * IDIM;

  __shared__ unsigned s_a[32 * 16];
  v8f c00{}, c01{}, c10{}, c11{};

  for (int k0 = 0; k0 < IS; k0 += 32) {
    int idx = threadIdx.x;
#pragma unroll
    for (int r = 0; r < 4; ++r, idx += 128) {
      const int m = idx >> 4, j = idx & 15;
      const unsigned* sr =
          (const unsigned*)sbuf + ((size_t)(t0 + m) * IS + k0) / 2;
      s_a[idx] = sr[j];
    }
    __syncthreads();
    v16bf a0 = load_a_lds(s_a);
    v16bf a1 = load_a_lds(s_a + 256);
    const unsigned short* Bk = w_sdown + (size_t)k0 * HDIM;
    v16bf b0 = load_b_bf16(Bk, HDIM, nb);
    v16bf b1 = load_b_bf16(Bk, HDIM, nb + 16);
    c00 = wmma_bf16(a0, b0, c00);
    c01 = wmma_bf16(a0, b1, c01);
    c10 = wmma_bf16(a1, b0, c10);
    c11 = wmma_bf16(a1, b1, c11);
    __syncthreads();
  }

  const int rowoff = (lane >> 4) ? 8 : 0;
  const int col = lane & 15;
#pragma unroll
  for (int r = 0; r < 8; ++r) {
    const int t = t0 + rowoff + r;
    out[(size_t)t * HDIM + nb + col]             = c00[r];
    out[(size_t)t * HDIM + nb + 16 + col]        = c01[r];
    out[(size_t)(t + 16) * HDIM + nb + col]      = c10[r];
    out[(size_t)(t + 16) * HDIM + nb + 16 + col] = c11[r];
  }
}

// ---------------- weighted combine of routed outputs -----------------------
__global__ void moe_combine_kernel(float* __restrict__ out,
                                   const unsigned short* __restrict__ ybuf,
                                   const float* __restrict__ topk_w) {
  size_t i = (size_t)blockIdx.x * blockDim.x + threadIdx.x;
  if (i >= (size_t)T_TOK * HDIM) return;
  const int t = (int)(i / HDIM);
  const int h = (int)(i % HDIM);
  float acc = out[i];
#pragma unroll
  for (int k = 0; k < TOPK; ++k) {
    const int pair = t * TOPK + k;
    acc += topk_w[pair] * bf2f(ybuf[(size_t)pair * HDIM + h]);
  }
  out[i] = acc;
}

// ---------------- host launcher --------------------------------------------
extern "C" void kernel_launch(void* const* d_in, const int* in_sizes, int n_in,
                              void* d_out, int out_size, void* d_ws,
                              size_t ws_size, hipStream_t stream) {
  (void)in_sizes; (void)n_in; (void)out_size; (void)ws_size;
  const float* x     = (const float*)d_in[0];
  const float* gw    = (const float*)d_in[1];
  const float* bias  = (const float*)d_in[2];
  const float* w_gu  = (const float*)d_in[3];
  const float* w_dn  = (const float*)d_in[4];
  const float* w_sgu = (const float*)d_in[5];
  const float* w_sdn = (const float*)d_in[6];
  float* out = (float*)d_out;

  char* ws = (char*)d_ws;
  size_t off = 0;
  auto alloc = [&](size_t bytes) -> void* {
    void* p = ws + off;
    off = (off + bytes + 255) & ~(size_t)255;
    return p;
  };
  const size_t n_x    = (size_t)T_TOK * HDIM;
  const size_t n_wgu  = (size_t)NEXP * HDIM * 2 * IDIM;
  const size_t n_wdn  = (size_t)NEXP * IDIM * HDIM;
  const size_t n_wsgu = (size_t)HDIM * 4 * IDIM;
  const size_t n_wsdn = (size_t)2 * IDIM * HDIM;

  unsigned short* xb     = (unsigned short*)alloc(n_x * 2);
  unsigned short* wgu_b  = (unsigned short*)alloc(n_wgu * 2);
  unsigned short* wdn_b  = (unsigned short*)alloc(n_wdn * 2);
  unsigned short* wsgu_b = (unsigned short*)alloc(n_wsgu * 2);
  unsigned short* wsdn_b = (unsigned short*)alloc(n_wsdn * 2);
  int*   topk_idx        = (int*)alloc((size_t)T_TOK * TOPK * 4);
  float* topk_w          = (float*)alloc((size_t)T_TOK * TOPK * 4);
  int*   counts          = (int*)alloc(NEXP * 4);
  int*   pair_of_slot    = (int*)alloc((size_t)NEXP * CAP * 4);
  unsigned short* hbuf   = (unsigned short*)alloc((size_t)T_TOK * TOPK * IDIM * 2);
  unsigned short* ybuf   = (unsigned short*)alloc((size_t)T_TOK * TOPK * HDIM * 2);
  unsigned short* sbuf   = (unsigned short*)alloc((size_t)T_TOK * 2 * IDIM * 2);

  auto cvt = [&](const float* s, unsigned short* d, size_t n) {
    const size_t n4 = n / 4;
    moe_cvt_kernel<<<(unsigned)((n4 + 255) / 256), 256, 0, stream>>>(s, d, n4);
  };
  cvt(x, xb, n_x);
  cvt(w_gu, wgu_b, n_wgu);
  cvt(w_dn, wdn_b, n_wdn);
  cvt(w_sgu, wsgu_b, n_wsgu);
  cvt(w_sdn, wsdn_b, n_wsdn);
  moe_zero_counts_kernel<<<1, 32, 0, stream>>>(counts);

  moe_route_kernel<<<T_TOK, 32, 0, stream>>>(x, gw, bias, topk_idx, topk_w,
                                             counts, pair_of_slot);
  moe_expert_gu_kernel<<<dim3(IDIM / 128, CAP / 32, NEXP), 128, 0, stream>>>(
      xb, wgu_b, counts, pair_of_slot, hbuf);
  moe_expert_down_kernel<<<dim3(HDIM / 128, CAP / 32, NEXP), 128, 0, stream>>>(
      hbuf, wdn_b, counts, pair_of_slot, ybuf);
  moe_shared_gu_kernel<<<dim3((2 * IDIM) / 128, T_TOK / 32), 128, 0, stream>>>(
      xb, wsgu_b, sbuf);
  moe_shared_down_kernel<<<dim3(HDIM / 128, T_TOK / 32), 128, 0, stream>>>(
      sbuf, wsdn_b, out);
  const size_t nelem = (size_t)T_TOK * HDIM;
  moe_combine_kernel<<<(unsigned)((nelem + 255) / 256), 256, 0, stream>>>(
      out, ybuf, topk_w);
}
